// mlp_with_som_32710470926378
// MI455X (gfx1250) — matrix operand order repaired and checked by reading, not compiled, not run on hardware
//
#include <hip/hip_runtime.h>

#define D_   768
#define H0_  1536
#define H1_  768
#define BSZ  8
#define C_   64
#define LX   256
#define LY   256

typedef __attribute__((ext_vector_type(16))) _Float16 v16h;
typedef __attribute__((ext_vector_type(8)))  float    v8f;
typedef __attribute__((ext_vector_type(4)))  int      v4i;

// ---------------------------------------------------------------------------
// CDNA5 async global->LDS copy (ASYNCcnt path), with safe fallbacks
// ---------------------------------------------------------------------------
#if defined(__gfx1250__) && __has_builtin(__builtin_amdgcn_global_load_async_to_lds_b128)
#define HAVE_ASYNC_LDS 1
#endif

__device__ __forceinline__ void async_cp16(void* lds_dst, const void* gsrc) {
#if defined(HAVE_ASYNC_LDS)
  typedef __attribute__((address_space(1))) v4i* as1p;
  typedef __attribute__((address_space(3))) v4i* as3p;
  __builtin_amdgcn_global_load_async_to_lds_b128((as1p)gsrc, (as3p)lds_dst, 0, 0);
#else
  *(uint4*)lds_dst = *(const uint4*)gsrc;
#endif
}

__device__ __forceinline__ void wait_async_lds() {
#if defined(__gfx1250__)
#if __has_builtin(__builtin_amdgcn_s_wait_asynccnt)
  __builtin_amdgcn_s_wait_asynccnt(0);
#else
  asm volatile("s_wait_asynccnt 0x0" ::: "memory");
#endif
#endif
}

// ---------------------------------------------------------------------------
// WMMA helpers (CDNA5 wave32, V_WMMA_F32_16X16X32_F16)
// ---------------------------------------------------------------------------
__device__ __forceinline__ v8f wmma_f16(v16h a, v16h b, v8f c) {
  return __builtin_amdgcn_wmma_f32_16x16x32_f16(false, a, false, b, (short)0, c,
                                                false, false);
}

// A fragment: 16x32 f16 tile, row-major at s with leading dim ld (elements).
__device__ __forceinline__ v16h load_a_frag(const _Float16* s, int ld) {
  const int lane = threadIdx.x & 31;
  const int m = lane & 15;
  const int k0 = (lane < 16) ? 0 : 8;
  const _Float16* row = s + m * ld;
  v16h r;
#pragma unroll
  for (int i = 0; i < 8; ++i) r[i] = row[k0 + i];
#pragma unroll
  for (int i = 0; i < 8; ++i) r[8 + i] = row[16 + k0 + i];
  return r;
}

// B fragment: B is 32x16 (KxN); storage is B^T row-major: s[n*ld + k].
__device__ __forceinline__ v16h load_bT_frag(const _Float16* s, int ld) {
  const int lane = threadIdx.x & 31;
  const int n = lane & 15;
  const int k0 = (lane < 16) ? 0 : 8;
  const _Float16* row = s + n * ld + k0;
  v16h r;
#pragma unroll
  for (int i = 0; i < 16; ++i) r[i] = row[(i < 8) ? i : 8 + i];
  return r;
}

__device__ __forceinline__ float softplus_f(float v) {
  if (v > 20.0f) return v;
  if (v < -20.0f) return __expf(v);
  return __logf(1.0f + __expf(v));
}

// ---------------------------------------------------------------------------
// Stage 0: utility kernels
// ---------------------------------------------------------------------------
__global__ void zero_kernel(float* __restrict__ p, int n) {
  int i = blockIdx.x * blockDim.x + threadIdx.x;
  if (i < n) p[i] = 0.0f;
}

__global__ void cvt_f16_kernel(const float* __restrict__ src,
                               _Float16* __restrict__ dst, int n) {
  int i = blockIdx.x * blockDim.x + threadIdx.x;
  int stride = gridDim.x * blockDim.x;
  for (; i < n; i += stride) dst[i] = (_Float16)src[i];
}

// ---------------------------------------------------------------------------
// Stage 1: scores (WMMA f16, double-buffered async K-pipeline) + argmax
// one 8-wave block per (b,c)
// ---------------------------------------------------------------------------
__global__ __launch_bounds__(256) void score_argmax_kernel(
    const _Float16* __restrict__ xs_h, const _Float16* __restrict__ ys_h,
    int* __restrict__ idx_out) {
  __shared__ _Float16 sA[2][LX * 32];  // xs K-chunk, double buffered (2x16 KB)
  __shared__ _Float16 sB[2][64 * 32];  // ys strip K-chunk (2x4 KB)
  __shared__ float    sS[LX * 64];     // score strip (64 KB)
  __shared__ float    sBestV[LX];
  __shared__ int      sBestI[LX];

  const int tid = threadIdx.x;
  const int wave = tid >> 5;
  const int lane = tid & 31;
  const int bc = blockIdx.x;
  const int b = bc / C_;
  const _Float16* xsb = xs_h + (size_t)b * LX * D_;
  const _Float16* ysb = ys_h + (size_t)bc * LY * D_;

  sBestV[tid] = -3.4e38f;
  sBestI[tid] = 0;

  for (int ystrip = 0; ystrip < LY / 64; ++ystrip) {
    const int ybase = ystrip * 64;

    // issue async copies of one 32-wide K chunk into buffer `buf`
    auto issue_chunk = [&](int buf, int k0) {
      // xs: 256 rows x 32 halves = 1024 x 16B transfers, 4 per thread
#pragma unroll
      for (int j = 0; j < 4; ++j) {
        const int e = tid + j * 256;
        const int row = e >> 2;
        const int q = e & 3;
        async_cp16(&sA[buf][row * 32 + q * 8],
                   xsb + (size_t)row * D_ + k0 + q * 8);
      }
      // ys strip: 64 rows x 32 halves = 256 x 16B transfers, 1 per thread
      {
        const int row = tid >> 2;
        const int q = tid & 3;
        async_cp16(&sB[buf][row * 32 + q * 8],
                   ysb + (size_t)(ybase + row) * D_ + k0 + q * 8);
      }
    };

    v8f acc[2][4];
#pragma unroll
    for (int xi = 0; xi < 2; ++xi)
#pragma unroll
      for (int yi = 0; yi < 4; ++yi) acc[xi][yi] = {};

    int cur = 0;
    issue_chunk(cur, 0);
    for (int kc = 0; kc < D_ / 32; ++kc) {
      wait_async_lds();
      __syncthreads();
      if (kc + 1 < D_ / 32) issue_chunk(cur ^ 1, (kc + 1) * 32);
#pragma unroll
      for (int xi = 0; xi < 2; ++xi) {
        v16h a = load_a_frag(&sA[cur][(wave * 2 + xi) * 16 * 32], 32);
#pragma unroll
        for (int yi = 0; yi < 4; ++yi) {
          v16h bf = load_bT_frag(&sB[cur][yi * 16 * 32], 32);
          acc[xi][yi] = wmma_f16(a, bf, acc[xi][yi]);
        }
      }
      cur ^= 1;
    }
    __syncthreads();

    // spill f32 strip to LDS (C/D layout: VGPR r -> M = 8*(lane/16)+r)
    {
      const int n = lane & 15;
      const int mb = (lane < 16) ? 0 : 8;
#pragma unroll
      for (int xi = 0; xi < 2; ++xi)
#pragma unroll
        for (int yi = 0; yi < 4; ++yi)
#pragma unroll
          for (int r = 0; r < 8; ++r)
            sS[((wave * 2 + xi) * 16 + mb + r) * 64 + yi * 16 + n] =
                acc[xi][yi][r];
    }
    __syncthreads();

    // running argmax: thread t scans x-row t over this 64-wide strip
    {
      float bv = sBestV[tid];
      int bi = sBestI[tid];
#pragma unroll 4
      for (int y = 0; y < 64; ++y) {
        float v = sS[tid * 64 + y];
        if (v > bv) { bv = v; bi = ybase + y; }
      }
      sBestV[tid] = bv;
      sBestI[tid] = bi;
    }
    __syncthreads();
  }
  idx_out[(size_t)bc * LX + tid] = sBestI[tid];
}

// ---------------------------------------------------------------------------
// Stage 2: fused gather + 3-layer MLP + row reduction, 32 rows per block
// ---------------------------------------------------------------------------
__global__ __launch_bounds__(256) void mlp_fused_kernel(
    const _Float16* __restrict__ xs_h, const _Float16* __restrict__ ys_h,
    const int* __restrict__ idxbuf,
    const _Float16* __restrict__ W0h, const float* __restrict__ b0,
    const _Float16* __restrict__ W1h, const float* __restrict__ b1,
    const _Float16* __restrict__ W2h, const float* __restrict__ b2,
    float* __restrict__ out) {
  __shared__ _Float16 sH[32 * (2 * D_)];  // 96 KB (reused as a2 later)
  __shared__ _Float16 sA1[32 * H0_];      // 96 KB
  __shared__ float sRed[32];
  __shared__ int sIdx[32];

  const int tid = threadIdx.x;
  const int wave = tid >> 5;
  const int lane = tid & 31;
  const int n0 = blockIdx.x * 32;
  const int bc = n0 / LX;  // all 32 rows share one (b,c)
  const int b = bc / C_;
  const int x0 = n0 % LX;

  if (tid < 32) sIdx[tid] = idxbuf[(size_t)bc * LX + x0 + tid];
  __syncthreads();

  // build h = [xs_row | ys_selected_row] with async b128 copies (f16 source)
  {
    const int CH = D_ / 8;  // 96 x 16B transfers per 768-half row-half
    for (int e = tid; e < 32 * CH; e += 256) {
      const int r = e / CH;
      const int ck = e % CH;
      async_cp16(&sH[r * (2 * D_) + ck * 8],
                 xs_h + ((size_t)b * LX + x0 + r) * D_ + ck * 8);
      async_cp16(&sH[r * (2 * D_) + D_ + ck * 8],
                 ys_h + ((size_t)bc * LY + sIdx[r]) * D_ + ck * 8);
    }
  }
  wait_async_lds();
  __syncthreads();

  // ---- layer 0: [32 x 1536] @ W0^T -> softplus -> sA1 [32 x 1536] ----
  for (int ot = wave; ot < H0_ / 16; ot += 8) {
    v8f acc0 = {}, acc1 = {};
    const _Float16* wrow = W0h + (size_t)(ot * 16) * (2 * D_);
    for (int kc = 0; kc < (2 * D_) / 32; ++kc) {
      __builtin_prefetch(wrow + (kc + 4) * 32, 0, 1);
      v16h bf = load_bT_frag(wrow + kc * 32, 2 * D_);
      v16h a0 = load_a_frag(&sH[kc * 32], 2 * D_);
      v16h a1 = load_a_frag(&sH[16 * (2 * D_) + kc * 32], 2 * D_);
      acc0 = wmma_f16(a0, bf, acc0);
      acc1 = wmma_f16(a1, bf, acc1);
    }
    const int nn = lane & 15;
    const int mb = (lane < 16) ? 0 : 8;
    const float bias = b0[ot * 16 + nn];
#pragma unroll
    for (int r = 0; r < 8; ++r) {
      sA1[(mb + r) * H0_ + ot * 16 + nn] =
          (_Float16)softplus_f(acc0[r] + bias);
      sA1[(16 + mb + r) * H0_ + ot * 16 + nn] =
          (_Float16)softplus_f(acc1[r] + bias);
    }
  }
  __syncthreads();

  // ---- layer 1: [32 x 1536] @ W1^T -> softplus -> sA2 [32 x 768] ----
  _Float16* sA2 = sH;  // h is dead; overlay
  for (int ot = wave; ot < H1_ / 16; ot += 8) {
    v8f acc0 = {}, acc1 = {};
    const _Float16* wrow = W1h + (size_t)(ot * 16) * H0_;
    for (int kc = 0; kc < H0_ / 32; ++kc) {
      __builtin_prefetch(wrow + (kc + 4) * 32, 0, 1);
      v16h bf = load_bT_frag(wrow + kc * 32, H0_);
      v16h a0 = load_a_frag(&sA1[kc * 32], H0_);
      v16h a1 = load_a_frag(&sA1[16 * H0_ + kc * 32], H0_);
      acc0 = wmma_f16(a0, bf, acc0);
      acc1 = wmma_f16(a1, bf, acc1);
    }
    const int nn = lane & 15;
    const int mb = (lane < 16) ? 0 : 8;
    const float bias = b1[ot * 16 + nn];
#pragma unroll
    for (int r = 0; r < 8; ++r) {
      sA2[(mb + r) * H1_ + ot * 16 + nn] =
          (_Float16)softplus_f(acc0[r] + bias);
      sA2[(16 + mb + r) * H1_ + ot * 16 + nn] =
          (_Float16)softplus_f(acc1[r] + bias);
    }
  }
  __syncthreads();

  // ---- layer 2: per-row dot with W2 (768) + b2, block-sum -> out[bc] ----
  if (tid < 32) {
    float s = 0.0f;
    const _Float16* row = &sA2[tid * H1_];
    for (int j = 0; j < H1_; ++j) s += (float)row[j] * (float)W2h[j];
    sRed[tid] = s + b2[0];
  }
  __syncthreads();
  if (tid == 0) {
    float t = 0.0f;
#pragma unroll
    for (int r = 0; r < 32; ++r) t += sRed[r];
    atomicAdd(&out[bc], t);
  }
}

// ---------------------------------------------------------------------------
// launcher
// ---------------------------------------------------------------------------
extern "C" void kernel_launch(void* const* d_in, const int* in_sizes, int n_in,
                              void* d_out, int out_size, void* d_ws,
                              size_t ws_size, hipStream_t stream) {
  (void)in_sizes; (void)n_in; (void)out_size; (void)ws_size;
  const float* xs = (const float*)d_in[0];
  const float* ys = (const float*)d_in[1];
  const float* W0 = (const float*)d_in[2];
  const float* b0 = (const float*)d_in[3];
  const float* W1 = (const float*)d_in[4];
  const float* b1 = (const float*)d_in[5];
  const float* W2 = (const float*)d_in[6];
  const float* b2 = (const float*)d_in[7];
  float* out = (float*)d_out;

  // workspace layout: indices | f16 weights | f16 xs | f16 ys
  char* ws = (char*)d_ws;
  int* idxbuf = (int*)ws;
  size_t off = (size_t)BSZ * C_ * LX * sizeof(int);
  off = (off + 255) & ~(size_t)255;
  _Float16* W0h = (_Float16*)(ws + off);
  off += (size_t)H0_ * (2 * D_) * sizeof(_Float16);
  _Float16* W1h = (_Float16*)(ws + off);
  off += (size_t)H1_ * H0_ * sizeof(_Float16);
  _Float16* W2h = (_Float16*)(ws + off);
  off += (size_t)H1_ * sizeof(_Float16);
  off = (off + 255) & ~(size_t)255;
  _Float16* xs_h = (_Float16*)(ws + off);
  off += (size_t)BSZ * LX * D_ * sizeof(_Float16);
  _Float16* ys_h = (_Float16*)(ws + off);
  off += (size_t)BSZ * C_ * LY * D_ * sizeof(_Float16);

  const int n_xs = BSZ * LX * D_;
  const int n_ys = BSZ * C_ * LY * D_;

  zero_kernel<<<2, 256, 0, stream>>>(out, BSZ * C_);
  cvt_f16_kernel<<<1024, 256, 0, stream>>>(W0, W0h, H0_ * 2 * D_);
  cvt_f16_kernel<<<512, 256, 0, stream>>>(W1, W1h, H1_ * H0_);
  cvt_f16_kernel<<<4, 256, 0, stream>>>(W2, W2h, H1_);
  cvt_f16_kernel<<<2048, 256, 0, stream>>>(xs, xs_h, n_xs);
  cvt_f16_kernel<<<8192, 256, 0, stream>>>(ys, ys_h, n_ys);

  score_argmax_kernel<<<BSZ * C_, 256, 0, stream>>>(xs_h, ys_h, idxbuf);

  mlp_fused_kernel<<<(BSZ * C_ * LX) / 32, 256, 0, stream>>>(
      xs_h, ys_h, idxbuf, W0h, b0, W1h, b1, W2h, b2, out);
}
